// ObjectRenderer_26130581029076
// MI455X (gfx1250) — matrix operand order repaired and checked by reading, not compiled, not run on hardware
//
#include <hip/hip_runtime.h>
#include <hip/hip_bf16.h>

// ---------------- problem constants ----------------
#define NRAYS 256
#define SP 64
#define SI 32
#define S 96              // SP+SI samples per ray
#define NS (NRAYS * S)    // 24576 total samples (== tokens for block 2)
#define CIN 32
#define CH 128            // C
#define HEADS 8
#define DH 16
#define KV 8              // high-res K
#define G 32
#define G2 (G * G)
#define G3 (G * G * G)
#define OCC_THRESH 0.7f
#define TOK1 (NS * KV)    // 196608 tokens for block 1

typedef __attribute__((ext_vector_type(16))) _Float16 v16h;
typedef __attribute__((ext_vector_type(8)))  _Float16 v8h;
typedef __attribute__((ext_vector_type(8)))  float    v8f;

// ---------------- WMMA fragment helpers (wave32, gfx1250 layouts) ----------------
// A-matrix 16x32 f16 from row-major f16 (ld in halves): two contiguous 16B loads per lane.
static __device__ __forceinline__ v16h frag_a_h16(const _Float16* __restrict__ Sm, int ld,
                                                  int k0, int lane) {
  int m  = lane & 15;
  int kb = (lane & 16) ? 8 : 0;
  const _Float16* p = Sm + (size_t)m * ld + k0 + kb;
  v8h lo = *(const v8h*)(p);
  v8h hi = *(const v8h*)(p + 16);
  return __builtin_shufflevector(lo, hi, 0, 1, 2, 3, 4, 5, 6, 7, 8, 9, 10, 11, 12, 13, 14, 15);
}

// B-matrix fragments pre-packed per (ktile, ntile, lane): 16 contiguous halves (32B).
static __device__ __forceinline__ v16h frag_b_pk(const _Float16* __restrict__ P, int ntiles,
                                                 int kt, int nt, int lane) {
  const _Float16* p = P + ((((size_t)kt * ntiles + nt) * 32 + lane) << 4);
  v8h lo = *(const v8h*)(p);
  v8h hi = *(const v8h*)(p + 8);
  return __builtin_shufflevector(lo, hi, 0, 1, 2, 3, 4, 5, 6, 7, 8, 9, 10, 11, 12, 13, 14, 15);
}

static __device__ __forceinline__ v8f wmma_step(v16h a, v16h b, v8f c) {
  return __builtin_amdgcn_wmma_f32_16x16x32_f16(false, a, false, b, (short)0, c, false, false);
}

// C/D 16x16 f32: lane l in [0,16): col n=l, rows m=r (vgpr r); lane l+16: rows m=8+r.
template <bool ADD>
static __device__ __forceinline__ void store_c_f32(float* __restrict__ Dst, int ld, int n0,
                                                   v8f acc, const float* __restrict__ bias,
                                                   int lane) {
  int n  = n0 + (lane & 15);
  int mb = (lane & 16) ? 8 : 0;
  float bn = bias[n];
#pragma unroll
  for (int r = 0; r < 8; ++r) {
    int m = mb + r;
    float v = acc[r] + bn;
    if (ADD) Dst[(size_t)m * ld + n] += v;
    else     Dst[(size_t)m * ld + n]  = v;
  }
}

template <bool LRELU>
static __device__ __forceinline__ void store_c_h16(_Float16* __restrict__ Dst, int ld, int n0,
                                                   v8f acc, const float* __restrict__ bias,
                                                   int lane) {
  int n  = n0 + (lane & 15);
  int mb = (lane & 16) ? 8 : 0;
  float bn = bias[n];
#pragma unroll
  for (int r = 0; r < 8; ++r) {
    int m = mb + r;
    float v = acc[r] + bn;
    if (LRELU) v = v > 0.f ? v : 0.01f * v;
    Dst[(size_t)m * ld + n] = (_Float16)v;
  }
}

// ---------------- weight packing: f32 row-major [Kreal x N] -> f16 fragment order ---------
__global__ __launch_bounds__(256) void pack_b_kernel(const float* __restrict__ W,
                                                     _Float16* __restrict__ P, int N,
                                                     int Kreal, int Kpad) {
  int e = blockIdx.x * 256 + threadIdx.x;
  int ntiles = N >> 4;
  int total = Kpad * N;
  if (e >= total) return;
  int j    = e & 15;
  int lane = (e >> 4) & 31;
  int rest = e >> 9;
  int nt = rest % ntiles;
  int kt = rest / ntiles;
  int k = kt * 32 + ((lane & 16) ? 8 : 0) + (j & 7) + ((j & 8) ? 16 : 0);
  int n = nt * 16 + (lane & 15);
  P[e] = (k < Kreal) ? (_Float16)W[(size_t)k * N + n] : (_Float16)0.f;
}

// ---------------- small math helpers ----------------
static __device__ __forceinline__ float sigmoidf_(float x) { return 1.f / (1.f + expf(-x)); }

static __device__ void trilerp(const float* __restrict__ vol, int C, float cz, float cy, float cx,
                               float* __restrict__ out) {
  cz = fminf(fmaxf(cz, 0.f), (float)(G - 1));
  cy = fminf(fmaxf(cy, 0.f), (float)(G - 1));
  cx = fminf(fmaxf(cx, 0.f), (float)(G - 1));
  int z0 = (int)floorf(cz), y0 = (int)floorf(cy), x0 = (int)floorf(cx);
  int z1 = min(z0 + 1, G - 1), y1 = min(y0 + 1, G - 1), x1 = min(x0 + 1, G - 1);
  float wz = cz - z0, wy = cy - y0, wx = cx - x0;
  int b000 = z0 * G2 + y0 * G + x0, b001 = z0 * G2 + y0 * G + x1;
  int b010 = z0 * G2 + y1 * G + x0, b011 = z0 * G2 + y1 * G + x1;
  int b100 = z1 * G2 + y0 * G + x0, b101 = z1 * G2 + y0 * G + x1;
  int b110 = z1 * G2 + y1 * G + x0, b111 = z1 * G2 + y1 * G + x1;
  float w000 = (1 - wz) * (1 - wy) * (1 - wx), w001 = (1 - wz) * (1 - wy) * wx;
  float w010 = (1 - wz) * wy * (1 - wx),       w011 = (1 - wz) * wy * wx;
  float w100 = wz * (1 - wy) * (1 - wx),       w101 = wz * (1 - wy) * wx;
  float w110 = wz * wy * (1 - wx),             w111 = wz * wy * wx;
  for (int c = 0; c < C; ++c) {
    const float* v = vol + (size_t)c * G3;
    out[c] = v[b000] * w000 + v[b001] * w001 + v[b010] * w010 + v[b011] * w011 +
             v[b100] * w100 + v[b101] * w101 + v[b110] * w110 + v[b111] * w111;
  }
}

static __device__ __forceinline__ void grid_coords(const float* __restrict__ ofc,
                                                   const float* __restrict__ dim,
                                                   const float p[3], float* cz, float* cy, float* cx) {
  float po[3];
#pragma unroll
  for (int i = 0; i < 3; ++i)
    po[i] = ofc[i * 3 + 0] * p[0] + ofc[i * 3 + 1] * p[1] + ofc[i * 3 + 2] * p[2] + ofc[9 + i];
  float u0 = 0.5f + po[0] / dim[0];
  float u1 = 0.5f + po[1] / dim[1];
  float u2 = 0.5f + po[2] / dim[2];
  *cx = u0 * (float)(G - 1);
  *cy = u1 * (float)(G - 1);
  *cz = u2 * (float)(G - 1);
}

static __device__ float ray_box(const float* __restrict__ ofc, const float o[3], const float d[3],
                                const float dim[3]) {
  float tlo = -3.0e38f, thi = 3.0e38f;
#pragma unroll
  for (int i = 0; i < 3; ++i) {
    float oo = ofc[i * 3 + 0] * o[0] + ofc[i * 3 + 1] * o[1] + ofc[i * 3 + 2] * o[2] + ofc[9 + i];
    float dd = ofc[i * 3 + 0] * d[0] + ofc[i * 3 + 1] * d[1] + ofc[i * 3 + 2] * d[2];
    if (fabsf(dd) < 1e-9f) dd = 1e-9f;
    float inv  = 1.f / dd;
    float half = dim[i] * 0.5f;
    float t0 = (-half - oo) * inv, t1 = (half - oo) * inv;
    tlo = fmaxf(tlo, fminf(t0, t1));
    thi = fminf(thi, fmaxf(t0, t1));
  }
  bool valid = thi >= fmaxf(tlo, 0.f);
  return valid ? tlo : __builtin_inff();
}

// ---------------- K0: invert pose (affine 4x4) ----------------
__global__ void inv_kernel(const float* __restrict__ pose, float* __restrict__ ofc) {
  if (threadIdx.x != 0) return;
  float a[9], b[3];
  for (int i = 0; i < 3; ++i) {
    for (int j = 0; j < 3; ++j) a[i * 3 + j] = pose[i * 4 + j];
    b[i] = pose[i * 4 + 3];
  }
  float c00 = a[4] * a[8] - a[5] * a[7];
  float c01 = a[5] * a[6] - a[3] * a[8];
  float c02 = a[3] * a[7] - a[4] * a[6];
  float det = a[0] * c00 + a[1] * c01 + a[2] * c02;
  float id  = 1.f / det;
  float R[9];
  R[0] = c00 * id; R[1] = (a[2] * a[7] - a[1] * a[8]) * id; R[2] = (a[1] * a[5] - a[2] * a[4]) * id;
  R[3] = c01 * id; R[4] = (a[0] * a[8] - a[2] * a[6]) * id; R[5] = (a[2] * a[3] - a[0] * a[5]) * id;
  R[6] = c02 * id; R[7] = (a[1] * a[6] - a[0] * a[7]) * id; R[8] = (a[0] * a[4] - a[1] * a[3]) * id;
  for (int i = 0; i < 9; ++i) ofc[i] = R[i];
  for (int i = 0; i < 3; ++i)
    ofc[9 + i] = -(R[i * 3 + 0] * b[0] + R[i * 3 + 1] * b[1] + R[i * 3 + 2] * b[2]);
}

// ---------------- K1: per-ray sampling + interp + importance + sort ----------------
__global__ __launch_bounds__(128) void ray_kernel(
    const float* __restrict__ rays_o, const float* __restrict__ rays_d,
    const float* __restrict__ dimension, const float* __restrict__ voxel_probs,
    const float* __restrict__ feature_vol, const float* __restrict__ u_primary,
    const float* __restrict__ u_cat, const float* __restrict__ u_lam,
    const float* __restrict__ ofcG, float* __restrict__ wMask, float* __restrict__ wRdo,
    float* __restrict__ wT, float* __restrict__ wOcc, float* __restrict__ wP,
    float* __restrict__ wX) {
  int ray = blockIdx.x, tid = threadIdx.x;
  __shared__ float sOfc[12], sDim[3];
  __shared__ float st[S], socc[S], scdf[SP - 1];
  __shared__ float sp3[S * 3];
  __shared__ float sxf[S * CIN];
  if (tid < 12) sOfc[tid] = ofcG[tid];
  if (tid < 3)  sDim[tid] = dimension[tid];
  __syncthreads();

  float o[3] = {rays_o[ray * 3], rays_o[ray * 3 + 1], rays_o[ray * 3 + 2]};
  float d[3] = {rays_d[ray * 3], rays_d[ray * 3 + 1], rays_d[ray * 3 + 2]};
  float t_near = ray_box(sOfc, o, d, sDim);
  float sumdim = fabsf(sDim[0]) + fabsf(sDim[1]) + fabsf(sDim[2]);
  float tmax = t_near + 10.f * sumdim;
  float pm[3] = {o[0] + tmax * d[0], o[1] + tmax * d[1], o[2] + tmax * d[2]};
  float dn[3] = {-d[0], -d[1], -d[2]};
  float t_far = tmax - ray_box(sOfc, pm, dn, sDim);
  bool m = __builtin_isfinite(t_near) && __builtin_isfinite(t_far) && (t_far > t_near);
  float tn = m ? t_near : 0.f, tf = m ? t_far : 1.f;

  if (tid < SP) {
    float u = u_primary[ray * SP + tid];
    float t = tn + (tf - tn) * ((float)tid + u) / (float)SP;
    st[tid] = t;
    float p[3] = {o[0] + t * d[0], o[1] + t * d[1], o[2] + t * d[2]};
    sp3[tid * 3] = p[0]; sp3[tid * 3 + 1] = p[1]; sp3[tid * 3 + 2] = p[2];
    float cz, cy, cx;
    grid_coords(sOfc, sDim, p, &cz, &cy, &cx);
    float oc; trilerp(voxel_probs, 1, cz, cy, cx, &oc);
    socc[tid] = oc;
    trilerp(feature_vol, CIN, cz, cy, cx, &sxf[tid * CIN]);
  }
  __syncthreads();
  if (tid == 0) {
    float cum = 0.f;
    for (int i = 0; i < SP - 1; ++i) {
      float pr = socc[i] + socc[i + 1];
      pr = fmaxf(pr, 1e-12f);
      cum += pr; scdf[i] = cum;
    }
    float inv = 1.f / cum;
    for (int i = 0; i < SP - 1; ++i) scdf[i] *= inv;
    wMask[ray] = m ? 1.f : 0.f;
    for (int i = 0; i < 3; ++i)
      wRdo[ray * 3 + i] = sOfc[i * 3 + 0] * d[0] + sOfc[i * 3 + 1] * d[1] + sOfc[i * 3 + 2] * d[2];
  }
  __syncthreads();
  if (tid < SI) {
    float u = u_cat[ray * SI + tid];
    int ind = 0;
    for (int k = 0; k < SP - 1; ++k) ind += (u >= scdf[k]) ? 1 : 0;
    if (ind > SP - 2) ind = SP - 2;
    float lam = u_lam[ray * SI + tid];
    float t = lam * st[ind] + (1.f - lam) * st[ind + 1];
    int idx = SP + tid;
    st[idx] = t;
    float p[3] = {o[0] + t * d[0], o[1] + t * d[1], o[2] + t * d[2]};
    sp3[idx * 3] = p[0]; sp3[idx * 3 + 1] = p[1]; sp3[idx * 3 + 2] = p[2];
    float cz, cy, cx;
    grid_coords(sOfc, sDim, p, &cz, &cy, &cx);
    float oc; trilerp(voxel_probs, 1, cz, cy, cx, &oc);
    socc[idx] = oc;
    trilerp(feature_vol, CIN, cz, cy, cx, &sxf[idx * CIN]);
  }
  __syncthreads();
  if (tid < S) {
    float ti = st[tid];
    int rank = 0;
    for (int j = 0; j < S; ++j) {
      float tj = st[j];
      rank += (tj < ti || (tj == ti && j < tid)) ? 1 : 0;
    }
    size_t base = (size_t)ray * S + rank;
    wT[base] = ti;
    wOcc[base] = socc[tid];
    wP[base * 3] = sp3[tid * 3]; wP[base * 3 + 1] = sp3[tid * 3 + 1]; wP[base * 3 + 2] = sp3[tid * 3 + 2];
    for (int c = 0; c < CIN; ++c) wX[base * CIN + c] = sxf[tid * CIN + c];
  }
}

// ---------------- K2: sigma MLP + outputs t/alpha/occ ----------------
__global__ __launch_bounds__(256) void sigma_kernel(
    const float* __restrict__ wT, const float* __restrict__ wOcc, const float* __restrict__ wX,
    const float* __restrict__ dW1, const float* __restrict__ db1, const float* __restrict__ dW2,
    const float* __restrict__ db2, const float* __restrict__ wMask, float* __restrict__ wOccT,
    float* __restrict__ out) {
  int sg = blockIdx.x * 256 + threadIdx.x;
  if (sg >= NS) return;
  int ray = sg / S;
  float mk = wMask[ray];
  const float* x = wX + (size_t)sg * CIN;
  float occ = wOcc[sg];
  float s2 = db2[0];
  for (int c = 0; c < CH; ++c) {
    float a = db1[c];
    for (int k = 0; k < CIN; ++k) a += x[k] * dW1[k * CH + c];
    a += occ * dW1[CIN * CH + c];
    a = a > 0.f ? a : 0.01f * a;
    s2 += a * dW2[c];
  }
  float sig = sigmoidf_(s2);
  float ot = (occ < OCC_THRESH) ? 0.f : occ;
  wOccT[sg] = ot;
  out[sg]          = wT[sg] * mk;
  out[NS + sg]     = ot * sig * mk;
  out[2 * NS + sg] = ot * mk;
}

// ---------------- K3: high-res features ----------------
__global__ __launch_bounds__(256) void xhr_kernel(
    const float* __restrict__ wP, const float* __restrict__ fvhr, const float* __restrict__ ofc,
    const float* __restrict__ dim, float* __restrict__ wXhr) {
  int item = blockIdx.x * 256 + threadIdx.x;
  if (item >= TOK1) return;
  int sg = item >> 3, k = item & 7;
  float p[3] = {wP[(size_t)sg * 3], wP[(size_t)sg * 3 + 1], wP[(size_t)sg * 3 + 2]};
  float cz, cy, cx;
  grid_coords(ofc, dim, p, &cz, &cy, &cx);
  float v[6];
  trilerp(fvhr + (size_t)k * 6 * G3, 6, cz, cy, cx, v);
  for (int c = 0; c < 6; ++c) wXhr[(size_t)sg * 48 + k * 6 + c] = v[c];
}

// ---------------- K4: fused attention block 1 (2 sequences x 8 tokens = 16 rows) --------
static __device__ void ln16h(const float* __restrict__ X, _Float16* __restrict__ Y, int tid,
                             float* __restrict__ mu, float* __restrict__ rs) {
  if (tid < 16) {
    float s = 0.f;
    for (int c = 0; c < CH; ++c) s += X[tid * CH + c];
    float mean = s / CH, v = 0.f;
    for (int c = 0; c < CH; ++c) { float dv = X[tid * CH + c] - mean; v += dv * dv; }
    mu[tid] = mean;
    rs[tid] = rsqrtf(v / CH + 1e-5f);
  }
  __syncthreads();
  for (int idx = tid; idx < 16 * CH; idx += 256) {
    int t = idx >> 7;
    Y[idx] = (_Float16)((X[idx] - mu[t]) * rs[t]);
  }
}

__global__ __launch_bounds__(256) void attn1_kernel(
    const float* __restrict__ wX, const float* __restrict__ wXhr,
    const float* __restrict__ wOccT, const float* __restrict__ wRdo,
    const float* __restrict__ ofc,
    const _Float16* __restrict__ pRW0, const float* __restrict__ rb0,
    const _Float16* __restrict__ pQKV, const float* __restrict__ bqkv,
    const _Float16* __restrict__ pWO, const float* __restrict__ bo,
    const _Float16* __restrict__ pF1, const float* __restrict__ bf1,
    const _Float16* __restrict__ pF2, const float* __restrict__ bf2,
    float* __restrict__ wHbar) {
  int tid = threadIdx.x, lane = tid & 31, wave = tid >> 5;
  int seq0 = blockIdx.x * 2;

  __shared__ __align__(32) _Float16 sXXh[16 * 64];
  __shared__ float sX[16 * 128];
  __shared__ __align__(32) _Float16 sHh[16 * 128];
  __shared__ float sQKV[16 * 384];
  __shared__ __align__(32) _Float16 sFFh[16 * 256];  // attn-out (cols 0..127), later FFN mid
  __shared__ float sMu[16], sRs[16];

  // build 42-dim (padded to 64) features, f16
  for (int idx = tid; idx < 16 * 64; idx += 256) {
    int t = idx >> 6, c = idx & 63;
    int sg = seq0 + (t >> 3);
    int k = t & 7;
    int ray = sg / S;
    float v = 0.f;
    if (c < 32) v = wX[(size_t)sg * CIN + c];
    else if (c < 35) v = wXhr[(size_t)sg * 48 + k * 6 + (c - 32)];
    else if (c < 38) {
      int i = c - 35;
      const float* hr = wXhr + (size_t)sg * 48 + k * 6;
      v = ofc[i * 3 + 0] * hr[3] + ofc[i * 3 + 1] * hr[4] + ofc[i * 3 + 2] * hr[5];
    } else if (c < 41) v = wRdo[ray * 3 + (c - 38)];
    else if (c == 41) v = wOccT[sg];
    sXXh[idx] = (_Float16)v;
  }
  __syncthreads();

  // h = xx2 @ rW0 + rb0   (16x64pad @ 64x128)
  {
    v8f acc = {};
#pragma unroll
    for (int kt = 0; kt < 2; ++kt)
      acc = wmma_step(frag_a_h16(sXXh, 64, kt * 32, lane), frag_b_pk(pRW0, 8, kt, wave, lane), acc);
    store_c_f32<false>(sX, CH, wave * 16, acc, rb0, lane);
  }
  __syncthreads();
  ln16h(sX, sHh, tid, sMu, sRs);
  __syncthreads();
  // qkv = ln(x) @ Wqkv + b   (16x128 @ 128x384): hoist A fragments over 3 n-tiles
  {
    v16h a0 = frag_a_h16(sHh, CH, 0, lane);
    v16h a1 = frag_a_h16(sHh, CH, 32, lane);
    v16h a2 = frag_a_h16(sHh, CH, 64, lane);
    v16h a3 = frag_a_h16(sHh, CH, 96, lane);
#pragma unroll
    for (int i = 0; i < 3; ++i) {
      int nt = wave + i * 8;
      v8f acc = {};
      acc = wmma_step(a0, frag_b_pk(pQKV, 24, 0, nt, lane), acc);
      acc = wmma_step(a1, frag_b_pk(pQKV, 24, 1, nt, lane), acc);
      acc = wmma_step(a2, frag_b_pk(pQKV, 24, 2, nt, lane), acc);
      acc = wmma_step(a3, frag_b_pk(pQKV, 24, 3, nt, lane), acc);
      store_c_f32<false>(sQKV, 3 * CH, nt * 16, acc, bqkv, lane);
    }
  }
  __syncthreads();
  // attention within each 8-token sequence
  if (tid < 128) {
    int seq = tid >> 6, h = (tid >> 3) & 7, qi = tid & 7;
    int tb = seq * 8;
    const float* q = &sQKV[(tb + qi) * 384 + h * DH];
    float sc[8], mx = -1e30f;
    for (int j = 0; j < 8; ++j) {
      const float* kk = &sQKV[(tb + j) * 384 + CH + h * DH];
      float s = 0.f;
      for (int dd = 0; dd < DH; ++dd) s += q[dd] * kk[dd];
      s *= 0.25f;
      sc[j] = s;
      mx = fmaxf(mx, s);
    }
    float den = 0.f;
    for (int j = 0; j < 8; ++j) { sc[j] = expf(sc[j] - mx); den += sc[j]; }
    float ov[DH];
    for (int dd = 0; dd < DH; ++dd) ov[dd] = 0.f;
    for (int j = 0; j < 8; ++j) {
      float w = sc[j] / den;
      const float* vv = &sQKV[(tb + j) * 384 + 2 * CH + h * DH];
      for (int dd = 0; dd < DH; ++dd) ov[dd] += w * vv[dd];
    }
    for (int dd = 0; dd < DH; ++dd) sFFh[(tb + qi) * 256 + h * DH + dd] = (_Float16)ov[dd];
  }
  __syncthreads();
  // x += o @ Wo + bo
  {
    v8f acc = {};
#pragma unroll
    for (int kt = 0; kt < 4; ++kt)
      acc = wmma_step(frag_a_h16(sFFh, 256, kt * 32, lane), frag_b_pk(pWO, 8, kt, wave, lane), acc);
    store_c_f32<true>(sX, CH, wave * 16, acc, bo, lane);
  }
  __syncthreads();
  ln16h(sX, sHh, tid, sMu, sRs);
  __syncthreads();
  // ff = lrelu(ln(x) @ Wf1 + b)   (16x128 @ 128x256), f16 out; hoist A over 2 n-tiles
  {
    v16h a0 = frag_a_h16(sHh, CH, 0, lane);
    v16h a1 = frag_a_h16(sHh, CH, 32, lane);
    v16h a2 = frag_a_h16(sHh, CH, 64, lane);
    v16h a3 = frag_a_h16(sHh, CH, 96, lane);
#pragma unroll
    for (int i = 0; i < 2; ++i) {
      int nt = wave + i * 8;
      v8f acc = {};
      acc = wmma_step(a0, frag_b_pk(pF1, 16, 0, nt, lane), acc);
      acc = wmma_step(a1, frag_b_pk(pF1, 16, 1, nt, lane), acc);
      acc = wmma_step(a2, frag_b_pk(pF1, 16, 2, nt, lane), acc);
      acc = wmma_step(a3, frag_b_pk(pF1, 16, 3, nt, lane), acc);
      store_c_h16<true>(sFFh, 2 * CH, nt * 16, acc, bf1, lane);
    }
  }
  __syncthreads();
  // x += ff @ Wf2 + b   (16x256 @ 256x128)
  {
    v8f acc = {};
#pragma unroll
    for (int kt = 0; kt < 8; ++kt)
      acc = wmma_step(frag_a_h16(sFFh, 256, kt * 32, lane), frag_b_pk(pF2, 8, kt, wave, lane), acc);
    store_c_f32<true>(sX, CH, wave * 16, acc, bf2, lane);
  }
  __syncthreads();
  // mean over the 8 tokens of each sequence
  {
    int seq = tid >> 7, c = tid & 127;
    float s = 0.f;
    for (int t = 0; t < 8; ++t) s += sX[(seq * 8 + t) * CH + c];
    wHbar[(size_t)(seq0 + seq) * CH + c] = s * 0.125f;
  }
}

// ---------------- K5: row LayerNorm, f16 output ----------------
__global__ __launch_bounds__(256) void ln_kernel(const float* __restrict__ X,
                                                 _Float16* __restrict__ Y, int rows) {
  int r = blockIdx.x * 256 + threadIdx.x;
  if (r >= rows) return;
  const float* x = X + (size_t)r * CH;
  _Float16* y = Y + (size_t)r * CH;
  float s = 0.f;
  for (int c = 0; c < CH; ++c) s += x[c];
  float mu = s / CH, v = 0.f;
  for (int c = 0; c < CH; ++c) { float d = x[c] - mu; v += d * d; }
  float rs = rsqrtf(v / CH + 1e-5f);
  for (int c = 0; c < CH; ++c) y[c] = (_Float16)((x[c] - mu) * rs);
}

// ---------------- K6: generic WMMA GEMM, 2 n-tiles per wave ----------------
// C = act(A@Bpacked + bias [+resid]); ACT: 0=none 1=lrelu; RES residual; OUTH f16 out.
template <int ACT, bool RES, bool OUTH>
__global__ __launch_bounds__(256) void gemm_kernel(
    const _Float16* __restrict__ A, const _Float16* __restrict__ Bp,
    const float* __restrict__ bias, const float* __restrict__ resid,
    void* __restrict__ Cd, int M, int N, int K) {
  int lane = threadIdx.x & 31, wave = threadIdx.x >> 5;
  int pair = blockIdx.x * 8 + wave;
  int tn = N >> 4, tp = N >> 5;
  int total = (M >> 4) * tp;
  if (pair >= total) return;
  int mt = pair / tp, np = pair - mt * tp;
  int nt0 = np * 2, nt1 = nt0 + 1;
  const _Float16* Ab = A + (size_t)mt * 16 * K;
  v8f acc0 = {}, acc1 = {};
  for (int kt = 0; kt < (K >> 5); ++kt) {
    v16h a = frag_a_h16(Ab, K, kt * 32, lane);
    acc0 = wmma_step(a, frag_b_pk(Bp, tn, kt, nt0, lane), acc0);
    acc1 = wmma_step(a, frag_b_pk(Bp, tn, kt, nt1, lane), acc1);
  }
  int mb = (lane & 16) ? 8 : 0;
#pragma unroll
  for (int half = 0; half < 2; ++half) {
    v8f acc = half ? acc1 : acc0;
    int n = (half ? nt1 : nt0) * 16 + (lane & 15);
    float bn = bias[n];
#pragma unroll
    for (int r = 0; r < 8; ++r) {
      int m = mt * 16 + mb + r;
      float v = acc[r] + bn;
      if (RES) v += resid[(size_t)m * N + n];
      if (ACT == 1) v = v > 0.f ? v : 0.01f * v;
      if (OUTH) ((_Float16*)Cd)[(size_t)m * N + n] = (_Float16)v;
      else      ((float*)Cd)[(size_t)m * N + n] = v;
    }
  }
}

// ---------------- K7: block-2 attention over 96 tokens per ray ----------------
__global__ __launch_bounds__(128) void attn2_kernel(const float* __restrict__ qkv,
                                                    _Float16* __restrict__ o) {
  int ray = blockIdx.x, tid = threadIdx.x;
  __shared__ float sK[S * DH], sV[S * DH];
  for (int h = 0; h < HEADS; ++h) {
    __syncthreads();
    for (int idx = tid; idx < S * DH; idx += 128) {
      int j = idx >> 4, dd = idx & 15;
      size_t base = ((size_t)(ray * S + j)) * 384;
      sK[idx] = qkv[base + CH + h * DH + dd];
      sV[idx] = qkv[base + 2 * CH + h * DH + dd];
    }
    __syncthreads();
    if (tid < S) {
      int qi = tid;
      size_t qb = ((size_t)(ray * S + qi)) * 384 + h * DH;
      float q[DH];
      for (int dd = 0; dd < DH; ++dd) q[dd] = qkv[qb + dd];
      float mx = -1e30f;
      for (int j = 0; j < S; ++j) {
        float s = 0.f;
        for (int dd = 0; dd < DH; ++dd) s += q[dd] * sK[j * DH + dd];
        mx = fmaxf(mx, s * 0.25f);
      }
      float den = 0.f, ov[DH];
      for (int dd = 0; dd < DH; ++dd) ov[dd] = 0.f;
      for (int j = 0; j < S; ++j) {
        float s = 0.f;
        for (int dd = 0; dd < DH; ++dd) s += q[dd] * sK[j * DH + dd];
        float e = expf(s * 0.25f - mx);
        den += e;
        for (int dd = 0; dd < DH; ++dd) ov[dd] += e * sV[j * DH + dd];
      }
      float inv = 1.f / den;
      for (int dd = 0; dd < DH; ++dd)
        o[((size_t)(ray * S + qi)) * CH + h * DH + dd] = (_Float16)(ov[dd] * inv);
    }
  }
}

// ---------------- K8: radiance head ----------------
__global__ __launch_bounds__(256) void rad_kernel(const float* __restrict__ x2,
                                                  const float* __restrict__ fW,
                                                  const float* __restrict__ fb,
                                                  const float* __restrict__ wMask,
                                                  float* __restrict__ out) {
  int sg = blockIdx.x * 256 + threadIdx.x;
  if (sg >= NS) return;
  int ray = sg / S;
  float mk = wMask[ray];
  const float* h = x2 + (size_t)sg * CH;
  for (int c = 0; c < 3; ++c) {
    float s = fb[c];
    for (int k = 0; k < CH; ++k) s += h[k] * fW[k * 3 + c];
    out[3 * NS + (size_t)sg * 3 + c] = sigmoidf_(s) * mk;
  }
}

// ---------------- launch ----------------
extern "C" void kernel_launch(void* const* d_in, const int* in_sizes, int n_in,
                              void* d_out, int out_size, void* d_ws, size_t ws_size,
                              hipStream_t stream) {
  const float* rays_o    = (const float*)d_in[0];
  const float* rays_d    = (const float*)d_in[1];
  const float* pose      = (const float*)d_in[2];
  const float* dimension = (const float*)d_in[3];
  const float* vprobs    = (const float*)d_in[4];
  const float* fvol      = (const float*)d_in[5];
  const float* fvhr      = (const float*)d_in[6];
  const float* u_primary = (const float*)d_in[7];
  const float* u_cat     = (const float*)d_in[8];
  const float* u_lam     = (const float*)d_in[9];
  const float* dW1 = (const float*)d_in[10]; const float* db1 = (const float*)d_in[11];
  const float* dW2 = (const float*)d_in[12]; const float* db2 = (const float*)d_in[13];
  const float* rW0 = (const float*)d_in[14]; const float* rb0 = (const float*)d_in[15];
  const float* nf_qkv = (const float*)d_in[16]; const float* nf_qkvb = (const float*)d_in[17];
  const float* nf_o   = (const float*)d_in[18]; const float* nf_ob   = (const float*)d_in[19];
  const float* nf_f1  = (const float*)d_in[20]; const float* nf_f1b  = (const float*)d_in[21];
  const float* nf_f2  = (const float*)d_in[22]; const float* nf_f2b  = (const float*)d_in[23];
  const float* rt_qkv = (const float*)d_in[24]; const float* rt_qkvb = (const float*)d_in[25];
  const float* rt_o   = (const float*)d_in[26]; const float* rt_ob   = (const float*)d_in[27];
  const float* rt_f1  = (const float*)d_in[28]; const float* rt_f1b  = (const float*)d_in[29];
  const float* rt_f2  = (const float*)d_in[30]; const float* rt_f2b  = (const float*)d_in[31];
  const float* fW = (const float*)d_in[32]; const float* fb = (const float*)d_in[33];

  float* out = (float*)d_out;
  float* w = (float*)d_ws;
  size_t off = 0;
  auto alloc = [&](size_t n) { float* p = w + off; off += n; return p; };
  auto allocH = [&](size_t nHalves) { return (_Float16*)alloc((nHalves + 1) / 2); };
  float* wOfc  = alloc(16);
  float* wMask = alloc(NRAYS);
  float* wRdo  = alloc(NRAYS * 3);
  float* wT    = alloc(NS);
  float* wOcc  = alloc(NS);
  float* wOccT = alloc(NS);
  float* wP    = alloc((size_t)NS * 3);
  float* wX    = alloc((size_t)NS * CIN);
  float* wXhr  = alloc((size_t)NS * 48);
  float* wHbar = alloc((size_t)NS * CH);
  float* wQkv  = alloc((size_t)NS * 3 * CH);
  float* wX2   = alloc((size_t)NS * CH);
  _Float16* wHn = allocH((size_t)NS * CH);
  _Float16* wO  = allocH((size_t)NS * CH);
  _Float16* wFF = allocH((size_t)NS * 2 * CH);
  // packed f16 weights (fragment layout)
  _Float16* pRW0  = allocH(64 * CH);
  _Float16* pQKV1 = allocH((size_t)CH * 3 * CH);
  _Float16* pWO1  = allocH((size_t)CH * CH);
  _Float16* pF11  = allocH((size_t)CH * 2 * CH);
  _Float16* pF21  = allocH((size_t)2 * CH * CH);
  _Float16* pQKV2 = allocH((size_t)CH * 3 * CH);
  _Float16* pWO2  = allocH((size_t)CH * CH);
  _Float16* pF12  = allocH((size_t)CH * 2 * CH);
  _Float16* pF22  = allocH((size_t)2 * CH * CH);
  (void)ws_size; (void)n_in; (void)in_sizes; (void)out_size;

  auto packGrid = [](int total) { return (total + 255) / 256; };
  pack_b_kernel<<<packGrid(64 * CH), 256, 0, stream>>>(rW0, pRW0, CH, 42, 64);
  pack_b_kernel<<<packGrid(CH * 3 * CH), 256, 0, stream>>>(nf_qkv, pQKV1, 3 * CH, CH, CH);
  pack_b_kernel<<<packGrid(CH * CH), 256, 0, stream>>>(nf_o, pWO1, CH, CH, CH);
  pack_b_kernel<<<packGrid(CH * 2 * CH), 256, 0, stream>>>(nf_f1, pF11, 2 * CH, CH, CH);
  pack_b_kernel<<<packGrid(2 * CH * CH), 256, 0, stream>>>(nf_f2, pF21, CH, 2 * CH, 2 * CH);
  pack_b_kernel<<<packGrid(CH * 3 * CH), 256, 0, stream>>>(rt_qkv, pQKV2, 3 * CH, CH, CH);
  pack_b_kernel<<<packGrid(CH * CH), 256, 0, stream>>>(rt_o, pWO2, CH, CH, CH);
  pack_b_kernel<<<packGrid(CH * 2 * CH), 256, 0, stream>>>(rt_f1, pF12, 2 * CH, CH, CH);
  pack_b_kernel<<<packGrid(2 * CH * CH), 256, 0, stream>>>(rt_f2, pF22, CH, 2 * CH, 2 * CH);

  inv_kernel<<<1, 32, 0, stream>>>(pose, wOfc);
  ray_kernel<<<NRAYS, 128, 0, stream>>>(rays_o, rays_d, dimension, vprobs, fvol,
                                        u_primary, u_cat, u_lam, wOfc, wMask, wRdo,
                                        wT, wOcc, wP, wX);
  sigma_kernel<<<NS / 256, 256, 0, stream>>>(wT, wOcc, wX, dW1, db1, dW2, db2, wMask, wOccT, out);
  xhr_kernel<<<TOK1 / 256, 256, 0, stream>>>(wP, fvhr, wOfc, dimension, wXhr);
  attn1_kernel<<<NS / 2, 256, 0, stream>>>(wX, wXhr, wOccT, wRdo, wOfc,
                                           pRW0, rb0, pQKV1, nf_qkvb, pWO1, nf_ob,
                                           pF11, nf_f1b, pF21, nf_f2b, wHbar);
  // ---- block 2 over 96 tokens/ray ----
  ln_kernel<<<NS / 256, 256, 0, stream>>>(wHbar, wHn, NS);
  {
    int total = (NS / 16) * (3 * CH / 32);
    gemm_kernel<0, false, false><<<(total + 7) / 8, 256, 0, stream>>>(
        wHn, pQKV2, rt_qkvb, nullptr, wQkv, NS, 3 * CH, CH);
  }
  attn2_kernel<<<NRAYS, 128, 0, stream>>>(wQkv, wO);
  {
    int total = (NS / 16) * (CH / 32);
    gemm_kernel<0, true, false><<<(total + 7) / 8, 256, 0, stream>>>(
        wO, pWO2, rt_ob, wHbar, wX2, NS, CH, CH);
  }
  ln_kernel<<<NS / 256, 256, 0, stream>>>(wX2, wHn, NS);
  {
    int total = (NS / 16) * (2 * CH / 32);
    gemm_kernel<1, false, true><<<(total + 7) / 8, 256, 0, stream>>>(
        wHn, pF12, rt_f1b, nullptr, wFF, NS, 2 * CH, CH);
  }
  {
    int total = (NS / 16) * (CH / 32);
    gemm_kernel<0, true, false><<<(total + 7) / 8, 256, 0, stream>>>(
        wFF, pF22, rt_f2b, wX2, wX2, NS, CH, 2 * CH);
  }
  rad_kernel<<<NS / 256, 256, 0, stream>>>(wX2, fW, fb, wMask, out);
}
